// InputStem_33827162423917
// MI455X (gfx1250) — matrix operand order repaired and checked by reading, not compiled, not run on hardware
//
#include <hip/hip_runtime.h>

typedef __attribute__((ext_vector_type(16))) __bf16 v16bf;
typedef __attribute__((ext_vector_type(8)))  __bf16 v8bf;
typedef __attribute__((ext_vector_type(8)))  float  v8f;
typedef __attribute__((ext_vector_type(4)))  unsigned int u32x4;
typedef __attribute__((ext_vector_type(4)))  int i32x4;
typedef __attribute__((ext_vector_type(8)))  int i32x8;
typedef unsigned short u16;
typedef unsigned int   u32;
typedef unsigned long long u64;

__device__ __forceinline__ float bfu2f(u16 u){ u32 v = ((u32)u) << 16; return __builtin_bit_cast(float, v); }
__device__ __forceinline__ u16 f2bfu(float f){            // RNE (used for stored tensors)
  u32 u = __builtin_bit_cast(u32, f);
  u += 0x7fffu + ((u >> 16) & 1u);
  return (u16)(u >> 16);
}
__device__ __forceinline__ u16 f2bfu_t(float f){          // truncate (transient fragments)
  return (u16)(__builtin_bit_cast(u32, f) >> 16);
}
__device__ __forceinline__ __bf16 u2bf(u16 u){ return __builtin_bit_cast(__bf16, u); }

__device__ __forceinline__ v8f wmma_bf16(v16bf a, v16bf b, v8f c){
  return __builtin_amdgcn_wmma_f32_16x16x32_bf16(false, a, false, b, (short)0, c, false, false);
}

// A-fragment loader: lane holds k = lh*8 + {0..7} and lh*8 + {16..23}  -> two b128 loads
__device__ __forceinline__ v16bf ldA(const u16* p){
  v8bf lo = *(const v8bf*)p;
  v8bf hi = *(const v8bf*)(p + 16);
  return __builtin_shufflevector(lo, hi, 0,1,2,3,4,5,6,7,8,9,10,11,12,13,14,15);
}
// 16 contiguous bf16 (16B-aligned) -> two b128 loads
__device__ __forceinline__ v16bf ldP(const u16* p){
  v8bf lo = *(const v8bf*)p;
  v8bf hi = *(const v8bf*)(p + 8);
  return __builtin_shufflevector(lo, hi, 0,1,2,3,4,5,6,7,8,9,10,11,12,13,14,15);
}

__device__ __forceinline__ void tdm_load(u32x4 g0, i32x8 g1, i32x4 g2, i32x4 g3){
#if __clang_major__ >= 23
  i32x8 pad = {};
  __builtin_amdgcn_tensor_load_to_lds(g0, g1, g2, g3, pad, 0);
#else
  __builtin_amdgcn_tensor_load_to_lds(g0, g1, g2, g3, 0);
#endif
}

// ---------------- small utility kernels ----------------

__global__ void zero_f32(float* p, int n){
  int i = blockIdx.x * blockDim.x + threadIdx.x;
  if (i < n) p[i] = 0.f;
}

__global__ void cvt_f32_bf16(const float* __restrict__ s, u16* __restrict__ d, int n){
  int i = blockIdx.x * blockDim.x + threadIdx.x;
  if (i < n) d[i] = f2bfu(s[i]);
}

// src[(o*I+i)*27+t] -> dst[(i*O+o)*32+t], taps padded 27->32 with zeros
__global__ void pack_w(const float* __restrict__ src, u16* __restrict__ dst, int O, int I){
  int idx = blockIdx.x * blockDim.x + threadIdx.x;
  int tot = O * I * 32;
  if (idx >= tot) return;
  int t = idx & 31; int oi = idx >> 5; int o = oi % O; int i = oi / O;
  dst[idx] = (t < 27) ? f2bfu(src[(o * I + i) * 27 + t]) : (u16)0;
}

__global__ void bnfin_kernel(const float* __restrict__ sum, const float* __restrict__ sq,
                             const float* __restrict__ g, const float* __restrict__ b,
                             float* __restrict__ sc, float* __restrict__ sh, int C, float invN){
  int c = threadIdx.x;
  if (c < C){
    float mean = sum[c] * invN;
    float var  = sq[c]  * invN - mean * mean;
    float s = g[c] * rsqrtf(var + 1e-5f);
    sc[c] = s; sh[c] = b[c] - mean * s;
  }
}

// normalize f3 (raw bf16, 61^3 per (b,c)) -> f3n, BN3+ReLU applied, W padded 61->64 (16B-aligned rows)
__global__ void bn3apply_kernel(const u16* __restrict__ f3, u16* __restrict__ f3n,
                                const float* __restrict__ sc3, const float* __restrict__ sh3){
  int bc = blockIdx.z;
  int c = bc & 31;
  int i = blockIdx.x * blockDim.x + threadIdx.x;
  if (i >= 61 * 61 * 64) return;
  int wp = i & 63; int zy = i >> 6; int y = zy % 61; int z = zy / 61;
  float v = 0.f;
  if (wp < 61)
    v = fmaxf(fmaf(sc3[c], bfu2f(f3[(u64)bc * 226981 + ((u64)z * 61 + y) * 61 + wp]), sh3[c]), 0.f);
  f3n[(u64)bc * 238144 + i] = f2bfu(v);
}

// ---------------- conv1: 1->16 ch, 3x3x3, direct f32 (tiny FLOPs) ----------------

__global__ void conv1_kernel(const float* __restrict__ x, const float* __restrict__ w,
                             u16* __restrict__ f1, float* __restrict__ gsum, float* __restrict__ gsq){
  __shared__ float swt[16 * 27];
  __shared__ float ssum[16], ssq[16];
  int tid = threadIdx.x;
  for (int i = tid; i < 432; i += blockDim.x) swt[i] = w[i];
  if (tid < 16){ ssum[tid] = 0.f; ssq[tid] = 0.f; }
  __syncthreads();

  long long gi = (long long)blockIdx.x * blockDim.x + tid;
  const long long NV = 2LL * 126 * 126 * 126;
  if (gi < NV){
    long long r = gi;
    int xx = (int)(r % 126); r /= 126;
    int y  = (int)(r % 126); r /= 126;
    int z  = (int)(r % 126);
    int b  = (int)(r / 126);
    float acc[16];
    #pragma unroll
    for (int c = 0; c < 16; c++) acc[c] = 0.f;
    #pragma unroll
    for (int i = 0; i < 3; i++)
      #pragma unroll
      for (int j = 0; j < 3; j++){
        const float* row = x + (((u64)b * 128 + (z + i)) * 128 + (y + j)) * 128 + xx;
        float v0 = row[0], v1 = row[1], v2 = row[2];
        int t = i * 9 + j * 3;
        #pragma unroll
        for (int c = 0; c < 16; c++)
          acc[c] = fmaf(swt[c*27+t], v0, fmaf(swt[c*27+t+1], v1, fmaf(swt[c*27+t+2], v2, acc[c])));
      }
    #pragma unroll
    for (int c = 0; c < 16; c++){
      f1[((((u64)b * 16 + c) * 126 + z) * 126 + y) * 126 + xx] = f2bfu(acc[c]);
      atomicAdd(&ssum[c], acc[c]); atomicAdd(&ssq[c], acc[c] * acc[c]);
    }
  }
  __syncthreads();
  if (tid < 16){ atomicAdd(&gsum[tid], ssum[tid]); atomicAdd(&gsq[tid], ssq[tid]); }
}

// ---------------- conv2: 16->16 implicit GEMM, double-buffered async->LDS ----------------
// Per channel: async-stage raw tile, each thread normalizes the dwords it staged (BN1+ReLU),
// one barrier, prefetch next channel, WMMA on current. Fragments are pure LDS u16 loads.

__global__ void conv2_kernel(const u16* __restrict__ f1, const u16* __restrict__ w2p,
                             const float* __restrict__ sc1, const float* __restrict__ sh1,
                             u16* __restrict__ f2, float* __restrict__ gsum, float* __restrict__ gsq){
  __shared__ u16 tile[2][9][68];          // 34 dwords per row, 1224 B per buffer
  __shared__ float ssum[16], ssq[16];
  const int tid  = threadIdx.x;
  const int lane = tid & 31;
  const int wv   = tid >> 5;
  const int lh   = lane >> 4;
  const int ln   = lane & 15;
  const int iw0  = blockIdx.x * 64;
  const int h    = blockIdx.y;
  const int bz   = blockIdx.z;
  const int b    = bz / 124;
  const int d    = bz % 124;
  if (tid < 16){ ssum[tid] = 0.f; ssq[tid] = 0.f; }

  const u64 f1base = (u64)(size_t)f1;
  const u64 CH1 = 2ull * 126 * 126 * 126;          // channel stride in bytes
  const u32 ldsbase = (u32)(size_t)&tile[0][0][0];

  // precompute staging jobs (<=3 per thread): 9 rows x 33 dwords
  int njobs = 0;
  u32 widx[3]; u64 gbase[3];
  for (int j = tid; j < 297; j += 128){
    int r = j / 33, q = j % 33;
    int kd = r / 3, kh = r % 3;
    widx[njobs]  = (u32)(r * 34 + q);
    gbase[njobs] = f1base + 2ull * (((u64)b * 16 * 126 * 126 * 126)
                 + (u64)(d + kd) * 126 * 126 + (u64)(h + kh) * 126 + iw0) + 4ull * q;
    njobs++;
  }
  // stage channel 0 into buffer 0
  for (int jj = 0; jj < njobs; ++jj){
    u32 la = ldsbase + widx[jj] * 4u;
    u64 ga = gbase[jj];
    asm volatile("global_load_async_to_lds_b32 %0, %1, off" :: "v"(la), "v"(ga) : "memory");
  }

  v8f acc = {};
  for (int c = 0; c < 16; c++){
    const int cur = c & 1;
    asm volatile("s_wait_asynccnt 0" ::: "memory");
    // normalize own staged dwords in place (BN1 + ReLU)
    {
      float sc = sc1[c], sh = sh1[c];
      u32* tw = (u32*)&tile[cur][0][0];
      for (int jj = 0; jj < njobs; ++jj){
        u32 wv2 = tw[widx[jj]];
        float v0 = fmaxf(fmaf(sc, bfu2f((u16)wv2), sh), 0.f);
        float v1 = fmaxf(fmaf(sc, bfu2f((u16)(wv2 >> 16)), sh), 0.f);
        tw[widx[jj]] = (u32)f2bfu_t(v0) | ((u32)f2bfu_t(v1) << 16);
      }
    }
    __syncthreads();                      // data ready for all; all done reading next buffer
    if (c + 1 < 16){                      // prefetch next channel into other buffer
      u32 bofs = (c + 1) & 1 ? 1224u : 0u;
      for (int jj = 0; jj < njobs; ++jj){
        u32 la = ldsbase + bofs + widx[jj] * 4u;
        u64 ga = gbase[jj] + (u64)(c + 1) * CH1;
        asm volatile("global_load_async_to_lds_b32 %0, %1, off" :: "v"(la), "v"(ga) : "memory");
      }
    }
    // fragments
    v16bf a = ldP(w2p + (c * 16 + ln) * 32 + lh * 16);
    v16bf bf;
    #pragma unroll
    for (int e = 0; e < 16; e++){         // B: lane = n + 16*k[4]; e = k[3:0]
      int k = e + (lh << 4);
      u16 raw = 0;
      if (k < 27){
        int kd = k / 9, rr = k % 9, kh = rr / 3, kw = rr % 3;
        raw = tile[cur][kd * 3 + kh][wv * 16 + ln + kw];
      }
      bf[e] = u2bf(raw);
    }
    acc = wmma_bf16(a, bf, acc);
  }

  int wout = iw0 + wv * 16 + ln;
  if (wout < 124){
    #pragma unroll
    for (int vg = 0; vg < 8; vg++){       // C: m = vg + 8*lanehi, n = lane[3:0]
      int m = vg + (lh << 3);
      float v = acc[vg];
      f2[((((u64)b * 16 + m) * 124 + d) * 124 + h) * 124 + wout] = f2bfu(v);
      atomicAdd(&ssum[m], v); atomicAdd(&ssq[m], v * v);
    }
  }
  __syncthreads();
  if (tid < 16){ atomicAdd(&gsum[tid], ssum[tid]); atomicAdd(&gsq[tid], ssq[tid]); }
}

// ---------------- conv3: 16->32, stride 2, TDM-staged, double-buffered ----------------
// Wave 0 issues one tensor_load_to_lds per channel: 3D tile 124 x 3(kh) x 3(kd),
// strides 124 / 124^2 elements, 2232 B contiguous into LDS. s_wait_tensorcnt ping-pong.

__global__ void conv3_kernel(const u16* __restrict__ f2, const u16* __restrict__ w3p,
                             const float* __restrict__ sc2, const float* __restrict__ sh2,
                             u16* __restrict__ f3, float* __restrict__ gsum, float* __restrict__ gsq){
  __shared__ u16 tile[2][1116];           // [kd*3+kh][124], contiguous per buffer
  __shared__ float ssum[32], ssq[32];
  const int tid  = threadIdx.x;
  const int lane = tid & 31;
  const int wv   = tid >> 5;
  const int lh   = lane >> 4;
  const int ln   = lane & 15;
  const int oh   = blockIdx.y;
  const int bz   = blockIdx.z;
  const int b    = bz / 61;
  const int od   = bz % 61;
  if (tid < 32){ ssum[tid] = ssq[tid] = 0.f; }

  const u64 f2base = (u64)(size_t)f2;
  const u64 tile0  = f2base + 2ull * (((u64)(b * 16) * 124 + (u64)(2 * od)) * 124 + (u64)(2 * oh)) * 124;
  const u64 CH2    = 2ull * 124 * 124 * 124;

  // constant D# pieces
  i32x8 g1;
  g1[0] = 0x00010000;                  // workgroup_mask=0, data_size=1 (2B)
  g1[1] = (int)(124u << 16);           // tensor_dim0 lo16
  g1[2] = (int)(124u << 16);           // tensor_dim0 hi=0 | tensor_dim1 lo16
  g1[3] = (int)(124u << 16);           // tensor_dim1 hi=0 | tile_dim0=124
  g1[4] = (int)(3u | (3u << 16));      // tile_dim1=3, tile_dim2=3
  g1[5] = 124;                         // tensor_dim0_stride lo32
  g1[6] = (int)(15376u << 16);         // stride0 hi=0 | tensor_dim1_stride lo16 (124*124)
  g1[7] = 0;                           // tensor_dim1_stride hi32
  i32x4 g2; g2[0] = 124; g2[1] = 0; g2[2] = 0; g2[3] = 0;   // tensor_dim2, tile_dim3=0
  i32x4 g3 = {};

  if (wv == 0){                         // stage channel 0 into buffer 0
    u64 ga = tile0;
    u32x4 g0;
    g0[0] = 1u;                         // count=1
    g0[1] = (u32)(size_t)&tile[0][0];
    g0[2] = (u32)ga;
    g0[3] = ((u32)(ga >> 32) & 0x01FFFFFFu) | 0x80000000u;   // addr[56:32] | type=2
    tdm_load(g0, g1, g2, g3);
  }

  v8f acc0 = {}, acc1 = {};
  for (int c = 0; c < 16; c++){
    const int cur = c & 1;
    __syncthreads();                    // everyone done reading buffer (c+1)&1 from iter c-1
    if (wv == 0){
      if (c + 1 < 16){
        u64 ga = tile0 + (u64)(c + 1) * CH2;
        u32x4 g0;
        g0[0] = 1u;
        g0[1] = (u32)(size_t)&tile[(c + 1) & 1][0];
        g0[2] = (u32)ga;
        g0[3] = ((u32)(ga >> 32) & 0x01FFFFFFu) | 0x80000000u;
        tdm_load(g0, g1, g2, g3);
        __builtin_amdgcn_s_wait_tensorcnt(1);   // oldest (channel c) complete
      } else {
        __builtin_amdgcn_s_wait_tensorcnt(0);
      }
    }
    __syncthreads();                    // buffer cur is valid for all waves

    float sc = sc2[c], sh = sh2[c];
    v16bf a0 = ldP(w3p + (c * 32 + ln) * 32 + lh * 16);
    v16bf a1 = ldP(w3p + (c * 32 + 16 + ln) * 32 + lh * 16);
    v16bf bf;
    #pragma unroll
    for (int e = 0; e < 16; e++){
      int k = e + (lh << 4);
      float v = 0.f;
      if (k < 27){
        int kd = k / 9, rr = k % 9, kh = rr / 3, kw = rr % 3;
        int iw = 2 * (wv * 16 + ln) + kw;
        u16 raw = (iw < 124) ? tile[cur][(kd * 3 + kh) * 124 + iw] : (u16)0;
        v = fmaxf(fmaf(sc, bfu2f(raw), sh), 0.f);
      }
      bf[e] = u2bf(f2bfu_t(v));
    }
    acc0 = wmma_bf16(a0, bf, acc0);
    acc1 = wmma_bf16(a1, bf, acc1);
  }

  int wout = wv * 16 + ln;
  if (wout < 61){
    #pragma unroll
    for (int vg = 0; vg < 8; vg++){
      int m = vg + (lh << 3);
      float v0 = acc0[vg], v1 = acc1[vg];
      f3[((((u64)b * 32 + m)      * 61 + od) * 61 + oh) * 61 + wout] = f2bfu(v0);
      f3[((((u64)b * 32 + m + 16) * 61 + od) * 61 + oh) * 61 + wout] = f2bfu(v1);
      atomicAdd(&ssum[m], v0);      atomicAdd(&ssq[m], v0 * v0);
      atomicAdd(&ssum[m + 16], v1); atomicAdd(&ssq[m + 16], v1 * v1);
    }
  }
  __syncthreads();
  if (tid < 32){ atomicAdd(&gsum[tid], ssum[tid]); atomicAdd(&gsq[tid], ssq[tid]); }
}

// ---------------- projection GEMM: split-K over channels, 2 M-tiles per wave ----------------
// f3n is pre-normalized & padded: all loads are b128, inner loop = 6 loads + 2 WMMA.

__global__ void gemm_kernel(const u16* __restrict__ f3n, const u16* __restrict__ wcat,
                            float* __restrict__ out){
  const int lane = threadIdx.x & 31;
  const int lh   = lane >> 4;
  const int ln   = lane & 15;
  const int ntile = blockIdx.x, mt = blockIdx.y, c = blockIdx.z;

  const int row = ntile * 16 + ln;
  const int b  = row / 216;
  const int r  = row % 216;
  const int dz = r / 36, dy = (r / 6) % 6, dx = r % 6;
  const u16* fch = f3n + (u64)(b * 32 + c) * 238144 + ((u64)(dz * 8) * 61 + (u64)(dy * 8)) * 64 + dx * 8;
  const u16* w0 = wcat + (u64)(mt * 32 + ln) * 131072 + (u64)c * 4096 + lh * 16;
  const u16* w1 = w0 + 16ull * 131072;

  v8f acc0 = {}, acc1 = {};
  for (int ks = 0; ks < 128; ks++){
    if ((ks & 31) == 0){
      __builtin_prefetch(w0 + ks * 32 + 1024, 0, 0);
      __builtin_prefetch(w1 + ks * 32 + 1024, 0, 0);
    }
    int pl0 = ks * 32 + (lh << 4);                 // lane-half K base: kd/kh constant over e
    int kd = pl0 >> 8, kh = (pl0 >> 4) & 15;
    v16bf bfr = ldP(fch + (kd * 61 + kh) * 64);
    v16bf a0  = ldP(w0 + ks * 32);
    v16bf a1  = ldP(w1 + ks * 32);
    acc0 = wmma_bf16(a0, bfr, acc0);
    acc1 = wmma_bf16(a1, bfr, acc1);
  }
  #pragma unroll
  for (int vg = 0; vg < 8; vg++){
    int mm = mt * 32 + vg + (lh << 3);
    atomicAdd(&out[(u64)row * 160 + mm],      acc0[vg]);
    atomicAdd(&out[(u64)row * 160 + mm + 16], acc1[vg]);
  }
}

// ---------------- epilogue: LayerNorm(tokens)+bias, aux+bias, coords ----------------

__global__ void ln_aux_kernel(const float* __restrict__ gout,
                              const float* __restrict__ tokb, const float* __restrict__ auxb,
                              const float* __restrict__ lng, const float* __restrict__ lnb,
                              float* __restrict__ dout){
  __shared__ float red[128];
  __shared__ float red2[128];
  int row = blockIdx.x;
  int t = threadIdx.x;
  float v = gout[row * 160 + t] + tokb[t];
  red[t] = v; red2[t] = v * v;
  __syncthreads();
  for (int s = 64; s > 0; s >>= 1){
    if (t < s){ red[t] += red[t + s]; red2[t] += red2[t + s]; }
    __syncthreads();
  }
  float mean = red[0] * (1.f / 128.f);
  float var  = red2[0] * (1.f / 128.f) - mean * mean;
  float inv  = rsqrtf(var + 1e-5f);
  dout[(u64)row * 128 + t] = (v - mean) * inv * lng[t] + lnb[t];
  if (t < 32) dout[55296 + row * 32 + t] = gout[row * 160 + 128 + t] + auxb[t];
  if (t < 3){
    int rr = row % 216;
    int dz = rr / 36, dy = (rr / 6) % 6, dx = rr % 6;
    int idx = (t == 0) ? dz : ((t == 1) ? dy : dx);
    dout[55296 + 13824 + row * 3 + t] = ((float)idx * 8.f + 7.5f) / 60.f;
  }
}

// ---------------- host launch ----------------

extern "C" void kernel_launch(void* const* d_in, const int* in_sizes, int n_in,
                              void* d_out, int out_size, void* d_ws, size_t ws_size,
                              hipStream_t stream){
  (void)in_sizes; (void)n_in; (void)out_size; (void)ws_size;
  const float* x    = (const float*)d_in[0];
  const float* c1w  = (const float*)d_in[1];
  const float* bn1g = (const float*)d_in[2];
  const float* bn1b = (const float*)d_in[3];
  const float* c2w  = (const float*)d_in[4];
  const float* bn2g = (const float*)d_in[5];
  const float* bn2b = (const float*)d_in[6];
  const float* c3w  = (const float*)d_in[7];
  const float* bn3g = (const float*)d_in[8];
  const float* bn3b = (const float*)d_in[9];
  const float* tokw = (const float*)d_in[10];
  const float* tokb = (const float*)d_in[11];
  const float* auxw = (const float*)d_in[12];
  const float* auxb = (const float*)d_in[13];
  const float* lng  = (const float*)d_in[14];
  const float* lnb  = (const float*)d_in[15];
  float* out = (float*)d_out;
  char* ws = (char*)d_ws;

  const u64 n_f1  = 2ull * 16 * 126 * 126 * 126;
  const u64 n_f2  = 2ull * 16 * 124 * 124 * 124;
  const u64 n_f3  = 2ull * 32 * 61 * 61 * 61;
  const u64 n_f3n = 64ull * 238144;              // padded W: 61*61*64 per (b,c)
  u64 off = 0;
  u16* f1  = (u16*)(ws + off); off += ((n_f1 * 2 + 255) & ~255ull);
  u16* f2  = (u16*)(ws + off); off += ((n_f2 * 2 + 255) & ~255ull);
  u16* f3  = (u16*)(ws + off); off += ((n_f3 * 2 + 255) & ~255ull);
  u16* f3n = (u16*)(ws + off); off += ((n_f3n * 2 + 255) & ~255ull);
  u16* w2p = (u16*)(ws + off); off += ((16ull * 16 * 32 * 2 + 255) & ~255ull);
  u16* w3p = (u16*)(ws + off); off += ((16ull * 32 * 32 * 2 + 255) & ~255ull);
  float* gout  = (float*)(ws + off); off += 432ull * 160 * 4;
  float* stats = (float*)(ws + off); off += 512;
  float* bnp   = (float*)(ws + off); off += 512;
  u16* wcat = f1;                                // 42 MB, aliased onto f1 once conv2 retires it

  zero_f32<<<(69376 + 255) / 256, 256, 0, stream>>>(gout, 69376);
  pack_w<<<(16 * 16 * 32 + 255) / 256, 256, 0, stream>>>(c2w, w2p, 16, 16);
  pack_w<<<(32 * 16 * 32 + 255) / 256, 256, 0, stream>>>(c3w, w3p, 32, 16);

  long long nv1 = 2LL * 126 * 126 * 126;
  conv1_kernel<<<(unsigned)((nv1 + 255) / 256), 256, 0, stream>>>(x, c1w, f1, stats + 0, stats + 16);
  bnfin_kernel<<<1, 32, 0, stream>>>(stats + 0, stats + 16, bn1g, bn1b, bnp + 0, bnp + 16, 16, 1.f / (float)nv1);

  conv2_kernel<<<dim3(2, 124, 248), 128, 0, stream>>>(f1, w2p, bnp + 0, bnp + 16, f2, stats + 32, stats + 48);
  bnfin_kernel<<<1, 32, 0, stream>>>(stats + 32, stats + 48, bn2g, bn2b, bnp + 32, bnp + 48, 16,
                                     1.f / (float)(2LL * 124 * 124 * 124));

  // f1 dead: convert projection weights into its space
  cvt_f32_bf16<<<(16777216 + 255) / 256, 256, 0, stream>>>(tokw, wcat, 128 * 131072);
  cvt_f32_bf16<<<(4194304 + 255) / 256, 256, 0, stream>>>(auxw, wcat + 128ull * 131072, 32 * 131072);

  conv3_kernel<<<dim3(1, 61, 122), 128, 0, stream>>>(f2, w3p, bnp + 32, bnp + 48, f3, stats + 64, stats + 96);
  bnfin_kernel<<<1, 32, 0, stream>>>(stats + 64, stats + 96, bn3g, bn3b, bnp + 64, bnp + 96, 32,
                                     1.f / (float)(2LL * 61 * 61 * 61));

  bn3apply_kernel<<<dim3(931, 1, 64), 256, 0, stream>>>(f3, f3n, bnp + 64, bnp + 96);

  gemm_kernel<<<dim3(27, 5, 32), 32, 0, stream>>>(f3n, wcat, gout);
  ln_aux_kernel<<<432, 128, 0, stream>>>(gout, tokb, auxb, lng, lnb, out);
}
